// ScalingAttention_197568495699
// MI455X (gfx1250) — matrix operand order repaired and checked by reading, not compiled, not run on hardware
//
#include <hip/hip_runtime.h>
#include <hip/hip_bf16.h>

// ---------------------------------------------------------------------------
// CDNA5 feature detection (compile-safe: falls back to sync paths if absent)
// ---------------------------------------------------------------------------
#if defined(__gfx1250__) && __has_builtin(__builtin_amdgcn_global_load_async_to_lds_b128)
#define USE_ASYNC_LDS 1
#else
#define USE_ASYNC_LDS 0
#endif

#if defined(__gfx1250__) && __has_builtin(__builtin_amdgcn_tensor_load_to_lds)
#define USE_TDM 1
#else
#define USE_TDM 0
#endif

// ---------------------------------------------------------------------------
// Types
// ---------------------------------------------------------------------------
typedef __attribute__((ext_vector_type(16))) __bf16 bf16x16;
typedef __attribute__((ext_vector_type(8)))  __bf16 bf16x8;
typedef __attribute__((ext_vector_type(8)))  float  f32x8;
typedef __attribute__((ext_vector_type(4)))  float  f32x4;
typedef __attribute__((ext_vector_type(4)))  unsigned int u32x4;
typedef __attribute__((ext_vector_type(8)))  int i32x8;
typedef __attribute__((ext_vector_type(4)))  int i32x4;

#define DIM      1024
#define HEADS    16
#define HEAD_DIM 64
#define SEQ_TGT  2048
#define SEQ_SRC  1024
#define BATCH    4

#define M_Q   (BATCH * SEQ_TGT)   // 8192
#define M_KV  (BATCH * SEQ_SRC)   // 4096

__device__ __forceinline__ __bf16 f2bf(float f) {
  unsigned u = __builtin_bit_cast(unsigned, f);
  u += 0x7FFFu + ((u >> 16) & 1u);          // round-to-nearest-even
  unsigned short h = (unsigned short)(u >> 16);
  return __builtin_bit_cast(__bf16, h);
}

__device__ __forceinline__ bf16x16 cat16(bf16x8 lo, bf16x8 hi) {
  return __builtin_shufflevector(lo, hi, 0,1,2,3,4,5,6,7,8,9,10,11,12,13,14,15);
}

__device__ __forceinline__ f32x8 wmma_bf16(bf16x16 a, bf16x16 b, f32x8 c) {
  // (neg_a, A, neg_b, B, c_mod, C, reuse_a, reuse_b)
  return __builtin_amdgcn_wmma_f32_16x16x32_bf16(false, a, false, b, (short)0, c,
                                                 false, false);
}

// ---------------------------------------------------------------------------
// Async global->LDS (ASYNCcnt) helpers.
// Builtin signature (from clang diagnostic): first param is
// 'int __vector_size__(16) __device__*' (AS1), second is the AS3 LDS pointer.
// ---------------------------------------------------------------------------
#if USE_ASYNC_LDS
typedef int gccv4i __attribute__((vector_size(16)));
typedef __attribute__((address_space(1))) gccv4i as1_v4i;
typedef __attribute__((address_space(3))) gccv4i as3_v4i;
__device__ __forceinline__ void async_ld16(const __bf16* g, __bf16* l) {
  // 16B per lane, memory -> LDS, no VGPR data path
  __builtin_amdgcn_global_load_async_to_lds_b128(
      (as1_v4i*)(unsigned long long)g,
      (as3_v4i*)(unsigned int)(unsigned long long)l, 0, 0);
}
template<int N> __device__ __forceinline__ void wait_async() {
#if __has_builtin(__builtin_amdgcn_s_wait_asynccnt)
  __builtin_amdgcn_s_wait_asynccnt(N);
#else
  asm volatile("s_wait_asynccnt %0" :: "i"(N) : "memory");
#endif
}
#else
template<int N> __device__ __forceinline__ void wait_async() {}
#endif

// ---------------------------------------------------------------------------
// TDM (Tensor Data Mover, TENSORcnt): one descriptor loads a 64x64 bf16 tile
// into LDS with a 16B pad every 128B row (-> 144B stride = A_LDS elements).
// ---------------------------------------------------------------------------
#if USE_TDM
__device__ __forceinline__ void tdm_load_tile64(const __bf16* gsrc, __bf16* lds_dst) {
  unsigned long long ga = (unsigned long long)gsrc;
  unsigned la = (unsigned)(unsigned long long)lds_dst;   // aperture low bits = LDS addr
  u32x4 g0;
  g0[0] = 1u;                                            // count=1, user descriptor
  g0[1] = la;                                            // lds_addr
  g0[2] = (unsigned)(ga & 0xFFFFFFFFu);                  // global_addr[31:0]
  g0[3] = (unsigned)((ga >> 32) & 0x01FFFFFFu) | (2u << 30); // ga[56:32] | type=2
  i32x8 g1;
  g1[0] = (1 << 16)        // data_size = 2 bytes
        | (1 << 20)        // pad_enable
        | (4 << 22)        // pad_interval: 32 DWORDs (128B row)
        | (3 << 25);       // pad_amount:   4 DWORDs (16B pad)
  g1[1] = (64 << 16);      // tensor_dim0[15:0]=64 (bits 79:48 start here)
  g1[2] = (64 << 16);      // tensor_dim0 msbs=0 | tensor_dim1[15:0]=64
  g1[3] = (64 << 16);      // tensor_dim1 msbs=0 | tile_dim0=64
  g1[4] = 64;              // tile_dim1=64 rows | tile_dim2=0
  g1[5] = DIM;             // tensor_dim0_stride[31:0] = 1024 elements
  g1[6] = 0;               // stride msbs / tensor_dim1_stride
  g1[7] = 0;
  i32x4 z4 = {0, 0, 0, 0};
#if __clang_major__ >= 23
  i32x8 z8 = {0, 0, 0, 0, 0, 0, 0, 0};
  __builtin_amdgcn_tensor_load_to_lds(g0, g1, z4, z4, z8, 0);
#else
  __builtin_amdgcn_tensor_load_to_lds(g0, g1, z4, z4, 0);
#endif
}
__device__ __forceinline__ void wait_tensor0() {
  __builtin_amdgcn_s_wait_tensorcnt(0);
}
#endif

// ---------------------------------------------------------------------------
// 1) fp32 -> bf16 elementwise conversion (8 elems / thread)
// ---------------------------------------------------------------------------
__global__ __launch_bounds__(256) void cvt_f32_to_bf16(
    const float* __restrict__ in, __bf16* __restrict__ out, int n)
{
  int i = (blockIdx.x * 256 + threadIdx.x) * 8;
  if (i + 8 <= n) {
    f32x4 a = *(const f32x4*)(in + i);
    f32x4 b = *(const f32x4*)(in + i + 4);
    bf16x8 o;
#pragma unroll
    for (int e = 0; e < 4; ++e) { o[e] = f2bf(a[e]); o[e + 4] = f2bf(b[e]); }
    *(bf16x8*)(out + i) = o;
  }
}

// ---------------------------------------------------------------------------
// 2) GEMM: C[M,N] = A[M,K] @ W[N,K]^T   (torch Linear semantics, bf16 in)
//    WG tile 128x128, 8 waves (4x2), wave tile 32x64, K-step 32.
//    Double-buffered LDS; staging via async global->LDS when available.
// ---------------------------------------------------------------------------
#define G_LDS 40   // 32 + 8 pad (row = 80B, multiple of 16B)

template<bool OUT_BF16, bool ADD_BIAS>
__global__ __launch_bounds__(256) void gemm_xwt(
    const __bf16* __restrict__ A, const __bf16* __restrict__ W,
    void* __restrict__ Cout, const float* __restrict__ bias,
    int M, int N, int K)
{
  __shared__ __align__(16) __bf16 sA[2][128 * G_LDS];
  __shared__ __align__(16) __bf16 sB[2][128 * G_LDS];

  const int t    = threadIdx.x;
  const int lane = t & 31;
  const int w    = t >> 5;
  const int wm   = w & 3;        // 4 waves along M
  const int wn   = w >> 2;       // 2 waves along N
  const int l15  = lane & 15;
  const int lh   = lane >> 4;    // half-wave selector
  const int m0   = blockIdx.y * 128;
  const int n0   = blockIdx.x * 128;

  f32x8 acc[2][4] = {};

  // stage one 128x32 A tile + 128x32 W tile into LDS buffer `buf`
  auto stage = [&](int buf, int k0) {
#pragma unroll
    for (int qq = 0; qq < 2; ++qq) {
      int c  = t * 2 + qq;              // 0..511
      int r  = c >> 2;                  // 0..127
      int c8 = (c & 3) * 8;             // 0/8/16/24
#if USE_ASYNC_LDS
      async_ld16(A + (size_t)(m0 + r) * K + k0 + c8, &sA[buf][r * G_LDS + c8]);
      async_ld16(W + (size_t)(n0 + r) * K + k0 + c8, &sB[buf][r * G_LDS + c8]);
#else
      *(bf16x8*)(&sA[buf][r * G_LDS + c8]) =
          *(const bf16x8*)(A + (size_t)(m0 + r) * K + k0 + c8);
      *(bf16x8*)(&sB[buf][r * G_LDS + c8]) =
          *(const bf16x8*)(W + (size_t)(n0 + r) * K + k0 + c8);
      if (k0 + 32 < K) {
        __builtin_prefetch(A + (size_t)(m0 + r) * K + k0 + 32 + c8, 0, 1);
        __builtin_prefetch(W + (size_t)(n0 + r) * K + k0 + 32 + c8, 0, 1);
      }
#endif
    }
  };

  stage(0, 0);
  const int nStage = K >> 5;
  for (int it = 0; it < nStage; ++it) {
    const int cur = it & 1;
    if (it + 1 < nStage) {
      stage(cur ^ 1, (it + 1) << 5);   // prefetch next stage into other buffer
      wait_async<4>();                 // wave issued 4 async/stage; drain stage `it`
    } else {
      wait_async<0>();
    }
    __syncthreads();                   // stage `it` visible to all waves

    // ---- A fragments: lane holds row (m), K halves per lh ----
    bf16x16 af[2];
#pragma unroll
    for (int mt = 0; mt < 2; ++mt) {
      const __bf16* ar = &sA[cur][(wm * 32 + mt * 16 + l15) * G_LDS];
      bf16x8 lo = *(const bf16x8*)(ar + lh * 8);
      bf16x8 hi = *(const bf16x8*)(ar + 16 + lh * 8);
      af[mt] = cat16(lo, hi);
    }
    // ---- B fragments: lane holds col (n), K halves per lh ----
    bf16x16 bfm[4];
#pragma unroll
    for (int nt = 0; nt < 4; ++nt) {
      const __bf16* br = &sB[cur][(wn * 64 + nt * 16 + l15) * G_LDS + lh * 8];
      bfm[nt] = cat16(*(const bf16x8*)(br), *(const bf16x8*)(br + 16));
    }
#pragma unroll
    for (int mt = 0; mt < 2; ++mt)
#pragma unroll
      for (int nt = 0; nt < 4; ++nt)
        acc[mt][nt] = wmma_bf16(af[mt], bfm[nt], acc[mt][nt]);
    __syncthreads();                   // all reads of buf `cur` done
  }

  // ---- epilogue: C layout row = r + 8*lh, col = l15 ----
#pragma unroll
  for (int mt = 0; mt < 2; ++mt)
#pragma unroll
    for (int nt = 0; nt < 4; ++nt)
#pragma unroll
      for (int r = 0; r < 8; ++r) {
        int row = m0 + wm * 32 + mt * 16 + r + 8 * lh;
        int col = n0 + wn * 64 + nt * 16 + l15;
        float v = acc[mt][nt][r];
        if (OUT_BF16) {
          ((__bf16*)Cout)[(size_t)row * N + col] = f2bf(v);
        } else {
          if (ADD_BIAS) v += bias[col];
          ((float*)Cout)[(size_t)row * N + col] = v;
        }
      }
}

// ---------------------------------------------------------------------------
// 3) Flash attention with block-causal mask: allow iff (i>>1) >= j.
//    Grid: (qtile=16, head=16, batch=4). WG = 256 thr = 8 waves, 16 q-rows/wave.
//    K tile fetched by the TDM (one descriptor, padded rows) when available.
// ---------------------------------------------------------------------------
#define A_LDS 72   // 64 + 8 pad (row = 144B, multiple of 16B) == TDM pad layout

__global__ __launch_bounds__(256) void attn_kernel(
    const __bf16* __restrict__ Qb, const __bf16* __restrict__ Kb,
    const __bf16* __restrict__ Vb, float* __restrict__ ctx)
{
  __shared__ __align__(16) __bf16 sK [64 * A_LDS];        // K block  [j][d]
  __shared__ __align__(16) __bf16 sVT[64 * A_LDS];        // V block^T [d][j]
  __shared__ __align__(16) __bf16 sP [8][16 * A_LDS];     // per-wave P [m][j]

  const int t    = threadIdx.x;
  const int lane = t & 31;
  const int w    = t >> 5;
  const int l15  = lane & 15;
  const int lh   = lane >> 4;
  const int qt   = blockIdx.x;           // query tile (0..15)
  const int h    = blockIdx.y;           // head
  const int b    = blockIdx.z;           // batch

  const int    i0    = qt * 128 + w * 16;                       // wave's q rows
  const size_t qbase = (size_t)b * SEQ_TGT * DIM + (size_t)h * HEAD_DIM;
  const size_t kbase = (size_t)b * SEQ_SRC * DIM + (size_t)h * HEAD_DIM;

  // Q fragments (registers for whole kernel): A layout, lane row = i0 + l15,
  // K(d) halves by lh; two d-steps of 32.
  bf16x16 qf[2];
  {
    const __bf16* qrow = Qb + qbase + (size_t)(i0 + l15) * DIM;
#pragma unroll
    for (int s = 0; s < 2; ++s) {
      bf16x8 lo = *(const bf16x8*)(qrow + s * 32 + lh * 8);
      bf16x8 hi = *(const bf16x8*)(qrow + s * 32 + 16 + lh * 8);
      qf[s] = cat16(lo, hi);
    }
  }

  f32x8 oacc[4] = {};
  float mrow[8], lrow[8];
#pragma unroll
  for (int r = 0; r < 8; ++r) { mrow[r] = -1e30f; lrow[r] = 0.f; }

  const int nblk = qt + 1;   // block-causal: WG needs j <= qt*64+63
  for (int jb = 0; jb < nblk; ++jb) {
    const int j0 = jb * 64;

    // ---- stage K block: TDM descriptor (wave 0) / async / sync fallback ----
#if USE_TDM
    if (w == 0)
      tdm_load_tile64(Kb + kbase + (size_t)j0 * DIM, sK);
#else
#pragma unroll
    for (int qq = 0; qq < 2; ++qq) {
      int c  = t * 2 + qq;          // 0..511
      int r  = c >> 3;              // j row 0..63
      int c8 = (c & 7) * 8;         // d 0..56
      const __bf16* src = Kb + kbase + (size_t)(j0 + r) * DIM + c8;
#if USE_ASYNC_LDS
      async_ld16(src, &sK[r * A_LDS + c8]);
#else
      *(bf16x8*)(&sK[r * A_LDS + c8]) = *(const bf16x8*)src;
#endif
    }
#endif
    // ---- stage transposed V block (sync scatter; TDM cannot transpose) ----
#pragma unroll
    for (int qq = 0; qq < 2; ++qq) {
      int c  = t * 2 + qq;
      int r  = c >> 3;
      int c8 = (c & 7) * 8;
      bf16x8 vv = *(const bf16x8*)(Vb + kbase + (size_t)(j0 + r) * DIM + c8);
#pragma unroll
      for (int e = 0; e < 8; ++e) sVT[(c8 + e) * A_LDS + r] = vv[e];
    }
#if USE_TDM
    wait_tensor0();
#elif USE_ASYNC_LDS
    wait_async<0>();
#endif
    __syncthreads();

    // ---- S = Q @ K^T (contract over d=64, two 32-steps) ----
    f32x8 sacc[4] = {};
#pragma unroll
    for (int s = 0; s < 2; ++s)
#pragma unroll
      for (int nt = 0; nt < 4; ++nt) {
        const __bf16* kr = &sK[(nt * 16 + l15) * A_LDS + s * 32 + lh * 8];
        bf16x16 bk = cat16(*(const bf16x8*)(kr), *(const bf16x8*)(kr + 16));
        sacc[nt] = wmma_bf16(qf[s], bk, sacc[nt]);
      }

    // ---- scale + block-causal mask + online softmax ----
#pragma unroll
    for (int r = 0; r < 8; ++r) {
      const int irow = i0 + r + 8 * lh;
      float mx = -1e30f;
#pragma unroll
      for (int nt = 0; nt < 4; ++nt) {
        int j = j0 + nt * 16 + l15;
        float v = sacc[nt][r] * 0.125f;            // 1/sqrt(64)
        if ((irow >> 1) < j) v = -1e30f;           // masked
        sacc[nt][r] = v;
        mx = fmaxf(mx, v);
      }
#pragma unroll
      for (int off = 8; off >= 1; off >>= 1)       // 16-lane row groups
        mx = fmaxf(mx, __shfl_xor(mx, off, 32));
      float mnew = fmaxf(mrow[r], mx);
      float corr = __expf(mrow[r] - mnew);
      float ssum = 0.f;
#pragma unroll
      for (int nt = 0; nt < 4; ++nt) {
        float p = __expf(sacc[nt][r] - mnew);
        sacc[nt][r] = p;
        ssum += p;
      }
#pragma unroll
      for (int off = 8; off >= 1; off >>= 1)
        ssum += __shfl_xor(ssum, off, 32);
      lrow[r] = lrow[r] * corr + ssum;
      mrow[r] = mnew;
#pragma unroll
      for (int dt = 0; dt < 4; ++dt) oacc[dt][r] = oacc[dt][r] * corr;
    }

    // ---- spill P (C layout) to per-wave LDS as bf16 [m][j] ----
#pragma unroll
    for (int nt = 0; nt < 4; ++nt)
#pragma unroll
      for (int r = 0; r < 8; ++r)
        sP[w][(r + 8 * lh) * A_LDS + nt * 16 + l15] = f2bf(sacc[nt][r]);
    // intra-wave only: DS ops are in-order per wave, just drain the counter
    asm volatile("s_wait_dscnt 0x0" ::: "memory");

    // ---- O += P @ V (contract over j=64, two 32-steps) ----
#pragma unroll
    for (int s2 = 0; s2 < 2; ++s2) {
      const __bf16* pr = &sP[w][l15 * A_LDS];
      bf16x8 lo = *(const bf16x8*)(pr + s2 * 32 + lh * 8);
      bf16x8 hi = *(const bf16x8*)(pr + s2 * 32 + 16 + lh * 8);
      bf16x16 pa = cat16(lo, hi);
#pragma unroll
      for (int dt = 0; dt < 4; ++dt) {
        const __bf16* vr = &sVT[(dt * 16 + l15) * A_LDS + s2 * 32 + lh * 8];
        bf16x16 bv = cat16(*(const bf16x8*)(vr), *(const bf16x8*)(vr + 16));
        oacc[dt] = wmma_bf16(pa, bv, oacc[dt]);
      }
    }
    __syncthreads();   // protect sK/sVT restage
  }

  // ---- normalize and write context (fp32, merged-head layout) ----
#pragma unroll
  for (int dt = 0; dt < 4; ++dt)
#pragma unroll
    for (int r = 0; r < 8; ++r) {
      int irow = i0 + r + 8 * lh;
      int d    = dt * 16 + l15;
      ctx[qbase + (size_t)irow * DIM + d] = oacc[dt][r] / lrow[r];
    }
}

// ---------------------------------------------------------------------------
// 4) LayerNorm over last dim (1024), bf16 output for final GEMM.
// ---------------------------------------------------------------------------
__global__ __launch_bounds__(256) void layernorm_bf16_out(
    const float* __restrict__ x, const float* __restrict__ gam,
    const float* __restrict__ bet, __bf16* __restrict__ out)
{
  __shared__ float red[16];
  const int row = blockIdx.x;
  const int t   = threadIdx.x;
  const float* xr = x + (size_t)row * DIM;
  f32x4 v = *(const f32x4*)(xr + t * 4);
  float s = v[0] + v[1] + v[2] + v[3];
  float q = v[0]*v[0] + v[1]*v[1] + v[2]*v[2] + v[3]*v[3];
#pragma unroll
  for (int off = 16; off >= 1; off >>= 1) {
    s += __shfl_xor(s, off, 32);
    q += __shfl_xor(q, off, 32);
  }
  const int w = t >> 5, lane = t & 31;
  if (lane == 0) { red[w] = s; red[8 + w] = q; }
  __syncthreads();
  if (t == 0) {
    float ts = 0.f, tq = 0.f;
    for (int i = 0; i < 8; ++i) { ts += red[i]; tq += red[8 + i]; }
    red[0] = ts; red[8] = tq;
  }
  __syncthreads();
  const float mean = red[0] * (1.0f / DIM);
  const float var  = red[8] * (1.0f / DIM) - mean * mean;
  const float rstd = rsqrtf(var + 1e-5f);
  f32x4 g  = *(const f32x4*)(gam + t * 4);
  f32x4 bb = *(const f32x4*)(bet + t * 4);
  __bf16* orow = out + (size_t)row * DIM + t * 4;
#pragma unroll
  for (int e = 0; e < 4; ++e)
    orow[e] = f2bf((v[e] - mean) * rstd * g[e] + bb[e]);
}

// ---------------------------------------------------------------------------
// Launch
// ---------------------------------------------------------------------------
extern "C" void kernel_launch(void* const* d_in, const int* in_sizes, int n_in,
                              void* d_out, int out_size, void* d_ws, size_t ws_size,
                              hipStream_t stream) {
  (void)in_sizes; (void)n_in; (void)out_size;
  const float* q    = (const float*)d_in[0];
  const float* kv   = (const float*)d_in[1];
  const float* Wq   = (const float*)d_in[2];
  const float* Wk   = (const float*)d_in[3];
  const float* Wv   = (const float*)d_in[4];
  const float* Wo   = (const float*)d_in[5];
  const float* bo   = (const float*)d_in[6];
  const float* ln_g = (const float*)d_in[7];
  const float* ln_b = (const float*)d_in[8];
  float* out = (float*)d_out;

  // workspace layout (bytes)
  char* ws = (char*)d_ws;
  const size_t SZ_QB  = (size_t)M_Q  * DIM * 2;   // 16 MiB  bf16 q
  const size_t SZ_KVB = (size_t)M_KV * DIM * 2;   //  8 MiB  bf16 kv
  const size_t SZ_WB  = (size_t)DIM  * DIM * 2;   //  2 MiB  bf16 weight
  const size_t SZ_CTX = (size_t)M_Q * DIM * 4;    // 32 MiB  fp32 context
  size_t off = 0;
  __bf16* qb  = (__bf16*)(ws + off); off += SZ_QB;
  __bf16* kvb = (__bf16*)(ws + off); off += SZ_KVB;
  __bf16* wqb = (__bf16*)(ws + off); off += SZ_WB;
  __bf16* wkb = (__bf16*)(ws + off); off += SZ_WB;
  __bf16* wvb = (__bf16*)(ws + off); off += SZ_WB;
  __bf16* wob = (__bf16*)(ws + off); off += SZ_WB;
  __bf16* Qb  = (__bf16*)(ws + off); off += SZ_QB;
  __bf16* Kb  = (__bf16*)(ws + off); off += SZ_KVB;
  __bf16* Vb  = (__bf16*)(ws + off); off += SZ_KVB;
  float*  ctx = (float*) (ws + off); off += SZ_CTX;
  __bf16* lnb = (__bf16*)(ws + off); off += SZ_QB;
  if (ws_size < off) return;   // not enough scratch

  const dim3 blk(256);

  // 1) bf16 conversion
  cvt_f32_to_bf16<<<(M_Q  * DIM) / 2048, blk, 0, stream>>>(q,  qb,  M_Q  * DIM);
  cvt_f32_to_bf16<<<(M_KV * DIM) / 2048, blk, 0, stream>>>(kv, kvb, M_KV * DIM);
  cvt_f32_to_bf16<<<(DIM * DIM) / 2048, blk, 0, stream>>>(Wq, wqb, DIM * DIM);
  cvt_f32_to_bf16<<<(DIM * DIM) / 2048, blk, 0, stream>>>(Wk, wkb, DIM * DIM);
  cvt_f32_to_bf16<<<(DIM * DIM) / 2048, blk, 0, stream>>>(Wv, wvb, DIM * DIM);
  cvt_f32_to_bf16<<<(DIM * DIM) / 2048, blk, 0, stream>>>(Wo, wob, DIM * DIM);

  // 2) QKV projections: X @ W^T  (bf16 out)
  gemm_xwt<true,  false><<<dim3(DIM / 128, M_Q  / 128), blk, 0, stream>>>(
      qb,  wqb, (void*)Qb, nullptr, M_Q,  DIM, DIM);
  gemm_xwt<true,  false><<<dim3(DIM / 128, M_KV / 128), blk, 0, stream>>>(
      kvb, wkb, (void*)Kb, nullptr, M_KV, DIM, DIM);
  gemm_xwt<true,  false><<<dim3(DIM / 128, M_KV / 128), blk, 0, stream>>>(
      kvb, wvb, (void*)Vb, nullptr, M_KV, DIM, DIM);

  // 3) flash attention with block-causal mask
  attn_kernel<<<dim3(SEQ_TGT / 128, HEADS, BATCH), blk, 0, stream>>>(Qb, Kb, Vb, ctx);

  // 4) layernorm -> bf16
  layernorm_bf16_out<<<M_Q, blk, 0, stream>>>(ctx, ln_g, ln_b, lnb);

  // 5) output projection: LN @ Wo^T + bo (fp32 out)
  gemm_xwt<false, true><<<dim3(DIM / 128, M_Q / 128), blk, 0, stream>>>(
      lnb, wob, (void*)out, bo, M_Q, DIM, DIM);
}
